// EfficientMultiHeadAttention_85358180041115
// MI455X (gfx1250) — matrix operand order repaired and checked by reading, compiled, not run on hardware
//
#include <hip/hip_runtime.h>
#include <hip/hip_bf16.h>
#include <math.h>

#define EMBED 384
#define NHEAD 6
#define QKVD  192
#define DHEAD 32
#define BATCH 4
#define SEQ   4096
#define MROWS (BATCH*SEQ)   /* 16384 */
#define ATT_SCALE 0.125f    /* (384/6)^-0.5 = 64^-0.5 */

typedef __bf16 bf16;
typedef __attribute__((ext_vector_type(16))) __bf16 v16bf;
typedef __attribute__((ext_vector_type(8)))  __bf16 v8bf;
typedef __attribute__((ext_vector_type(8)))  float  v8f;

static __device__ __forceinline__ v16bf cat8(v8bf a, v8bf b) {
    return __builtin_shufflevector(a, b, 0,1,2,3,4,5,6,7,8,9,10,11,12,13,14,15);
}
static __device__ __forceinline__ v8f wmma_bf16(v16bf a, v16bf b, v8f c) {
    // D = A(16x32 bf16) * B(32x16 bf16) + C(16x16 f32)
    return __builtin_amdgcn_wmma_f32_16x16x32_bf16(false, a, false, b, (short)0, c,
                                                   false, false);
}

// ---------------------------------------------------------------------------
// Kernel 0a: convert x to bf16 once (each element reused 36x by QKV GEMM).
// 4 elements / thread: float4 load -> 2x v_cvt_pk_bf16_f32 -> 8B store.
// ---------------------------------------------------------------------------
__global__ void __launch_bounds__(256)
conv_x(const float* __restrict__ x, bf16* __restrict__ xb) {
    size_t i = ((size_t)blockIdx.x*256 + threadIdx.x) * 4;   // < 16384*384
    float4 v = *(const float4*)(x + i);
    bf16 o[4] = { (bf16)v.x, (bf16)v.y, (bf16)v.z, (bf16)v.w };
    *(uint2*)(xb + i) = *(uint2*)o;
}

// ---------------------------------------------------------------------------
// Kernel 0b: weights -> bf16, transposed to [N][K] so the WMMA B-operand
// (lane = column n, 16 contiguous K halves split at lane 16) is two B128s.
// ---------------------------------------------------------------------------
__global__ void __launch_bounds__(256)
prep_weights(const float* __restrict__ qkv_w, const float* __restrict__ proj_w,
             bf16* __restrict__ w1t, bf16* __restrict__ w2t) {
    const int N1 = (3*QKVD)*EMBED;   // 576*384
    const int N2 = EMBED*QKVD;       // 384*192
    int tid = blockIdx.x*256 + threadIdx.x;
    if (tid < N1) {
        int n = tid / EMBED, k = tid % EMBED;
        w1t[tid] = (bf16)qkv_w[(size_t)k*(3*QKVD) + n];
    } else if (tid < N1 + N2) {
        int j = tid - N1;
        int n = j / QKVD, k = j % QKVD;
        w2t[j] = (bf16)proj_w[(size_t)k*EMBED + n];
    }
}

// ---------------------------------------------------------------------------
// Kernel 1: QKV GEMM [16384,384] x [384,576] + bias.  One wave computes a
// 16x64 strip (4 accumulators, A reused across 4 B tiles): per K-step
// 2 A-loads + 8 B-loads + 4 WMMAs.  K loop = 12 steps = 48 WMMAs / wave.
// Epilogue scatters q,k row-major [bh][n][d] and v transposed [bh][d][n].
// ---------------------------------------------------------------------------
__global__ void __launch_bounds__(256)
qkv_gemm(const bf16* __restrict__ xb, const float* __restrict__ qkv_b,
         const bf16* __restrict__ w1t,
         bf16* __restrict__ qb, bf16* __restrict__ kb, bf16* __restrict__ vtb) {
    const int lane = threadIdx.x & 31;
    const int wid  = threadIdx.x >> 5;
    const int tile = blockIdx.x*8 + wid;       // 1024 M-tiles * 9 N-groups
    const int mt = tile / 9, ng = tile % 9;    // N-group = 64 columns
    const int g = lane >> 4, lr = lane & 15;

    const bf16* arow  = xb  + (size_t)(mt*16 + lr)*EMBED;
    const bf16* brow0 = w1t + (size_t)(ng*64 + lr)*EMBED;

    v8f acc[4] = {};
    #pragma unroll
    for (int kt = 0; kt < EMBED; kt += 32) {
        // A: lane = row m; halves = K {16g..16g+15}
        v16bf A = cat8(*(const v8bf*)(arow + kt + 16*g),
                       *(const v8bf*)(arow + kt + 16*g + 8));
        #pragma unroll
        for (int j = 0; j < 4; ++j) {          // reuse A across 4 B tiles
            const bf16* br = brow0 + (size_t)j*16*EMBED;
            v16bf B = cat8(*(const v8bf*)(br + kt + 16*g),
                           *(const v8bf*)(br + kt + 16*g + 8));
            acc[j] = wmma_bf16(A, B, acc[j]);
        }
    }

    // Epilogue: D layout -> (m = r + 8g, n = lr).  64-col group stays inside
    // one q/k/v chunk (192 = 3*64); head may change per 16-col sub-tile.
    #pragma unroll
    for (int j = 0; j < 4; ++j) {
        const int n0    = ng*64 + j*16;
        const int chunk = n0 / QKVD;            // 0=q 1=k 2=v
        const int h     = (n0 % QKVD) / DHEAD;
        const int dim   = (n0 % DHEAD) + lr;    // n0%32 is 0 or 16; dim < 32
        const float bias = qkv_b[n0 + lr];
        #pragma unroll
        for (int r = 0; r < 8; ++r) {
            int m  = mt*16 + r + 8*g;           // row in [0,16384)
            int b  = m >> 12, np = m & (SEQ-1);
            size_t bh = (size_t)b*NHEAD + h;
            float v = acc[j][r] + bias;
            if (chunk == 0)      qb[(bh*SEQ + np)*DHEAD + dim] = (bf16)v;
            else if (chunk == 1) kb[(bh*SEQ + np)*DHEAD + dim] = (bf16)v;
            else                 vtb[(bh*DHEAD + dim)*SEQ + np] = (bf16)v;
        }
    }
}

// ---------------------------------------------------------------------------
// Kernel 2: flash attention. 8 waves/WG; wave owns 16 query rows; WG stages
// 32-key K/V tiles in LDS. Online softmax: cross-lane reductions via
// __shfl_xor masks 1..8 (stay within each 16-lane half = C/D layout group).
// ---------------------------------------------------------------------------
__global__ void __launch_bounds__(256)
flash_attn(const bf16* __restrict__ qb, const bf16* __restrict__ kb,
           const bf16* __restrict__ vtb, bf16* __restrict__ att) {
    __shared__ bf16 ldsK[32][32];       // [key][dim]
    __shared__ bf16 ldsV[32][32];       // [dim][key]  (v stored transposed)
    __shared__ bf16 ldsP[8][16][32];    // per-wave P transpose scratch

    const int lane = threadIdx.x & 31;
    const int wid  = threadIdx.x >> 5;
    const int g = lane >> 4, lr = lane & 15;
    const int bh = blockIdx.x / 32;                 // 24 (b,h) pairs
    const int b  = bh / NHEAD, h = bh % NHEAD;
    const int qbase = (blockIdx.x % 32) * 128 + wid*16;

    // Q A-operand, resident in registers for the whole loop
    const bf16* qrow = qb + ((size_t)bh*SEQ + qbase + lr)*DHEAD;
    v16bf Aq = cat8(*(const v8bf*)(qrow + 16*g),
                    *(const v8bf*)(qrow + 16*g + 8));

    v8f acc0 = {}, acc1 = {};                       // out dims 0-15 / 16-31
    float mi[8], li[8];
    #pragma unroll
    for (int r = 0; r < 8; ++r) { mi[r] = -1e30f; li[r] = 0.f; }

    const int t   = threadIdx.x;
    const int krw = t >> 3;          // 0..31: row of K/V tile this thread loads
    const int cof = (t & 7) * 4;     // 4 bf16 (8B) per thread per tile
    const bf16* kbase = kb  + (size_t)bh*SEQ*DHEAD;
    const bf16* vbase = vtb + (size_t)bh*DHEAD*SEQ;

    for (int kt = 0; kt < SEQ; kt += 32) {
        // stage tiles (coalesced 8B per thread)
        *(uint2*)&ldsK[krw][cof] = *(const uint2*)(kbase + (size_t)(kt+krw)*DHEAD + cof);
        *(uint2*)&ldsV[krw][cof] = *(const uint2*)(vbase + (size_t)krw*SEQ + kt + cof);
        if (kt + 32 < SEQ) {        // global_prefetch_b8 for next tile
            __builtin_prefetch(kbase + (size_t)(kt+32+krw)*DHEAD + cof, 0, 1);
            __builtin_prefetch(vbase + (size_t)krw*SEQ + kt + 32 + cof, 0, 1);
        }
        __syncthreads();

        // S = q k^T : B lane = key column; 16 contiguous dim halves at 16g
        v16bf Bk0 = cat8(*(const v8bf*)&ldsK[lr][16*g],    *(const v8bf*)&ldsK[lr][16*g+8]);
        v16bf Bk1 = cat8(*(const v8bf*)&ldsK[16+lr][16*g], *(const v8bf*)&ldsK[16+lr][16*g+8]);
        v8f zero = {};
        v8f s0 = wmma_bf16(Aq, Bk0, zero);   // keys kt..kt+15
        v8f s1 = wmma_bf16(Aq, Bk1, zero);   // keys kt+16..kt+31

        float p0[8], p1[8];
        #pragma unroll
        for (int r = 0; r < 8; ++r) {
            float a = s0[r]*ATT_SCALE, c = s1[r]*ATT_SCALE;
            float rm = fmaxf(a, c);
            #pragma unroll
            for (int msk = 1; msk < 16; msk <<= 1)
                rm = fmaxf(rm, __shfl_xor(rm, msk, 32));
            float mn = fmaxf(mi[r], rm);
            float sc = __expf(mi[r] - mn);
            p0[r] = __expf(a - mn);
            p1[r] = __expf(c - mn);
            float rs = p0[r] + p1[r];
            #pragma unroll
            for (int msk = 1; msk < 16; msk <<= 1)
                rs += __shfl_xor(rs, msk, 32);
            li[r] = li[r]*sc + rs;
            mi[r] = mn;
            acc0[r] = acc0[r]*sc;
            acc1[r] = acc1[r]*sc;
        }

        // transpose P (D layout -> A layout) through per-wave LDS scratch
        #pragma unroll
        for (int r = 0; r < 8; ++r) {
            ldsP[wid][r + 8*g][lr]      = (bf16)p0[r];
            ldsP[wid][r + 8*g][16 + lr] = (bf16)p1[r];
        }
        __syncthreads();
        v16bf Ap  = cat8(*(const v8bf*)&ldsP[wid][lr][16*g],
                         *(const v8bf*)&ldsP[wid][lr][16*g + 8]);
        // O += P V : B lane = out-dim column; 16 contiguous key halves at 16g
        v16bf Bv0 = cat8(*(const v8bf*)&ldsV[lr][16*g],    *(const v8bf*)&ldsV[lr][16*g+8]);
        v16bf Bv1 = cat8(*(const v8bf*)&ldsV[16+lr][16*g], *(const v8bf*)&ldsV[16+lr][16*g+8]);
        acc0 = wmma_bf16(Ap, Bv0, acc0);
        acc1 = wmma_bf16(Ap, Bv1, acc1);
        __syncthreads();   // before next tile overwrites ldsK/ldsV
    }

    // normalize and store to att [B][SEQ][192] (head-interleaved, bf16)
    #pragma unroll
    for (int r = 0; r < 8; ++r) {
        int row = qbase + r + 8*g;
        float inv = 1.f / li[r];
        size_t base = ((size_t)b*SEQ + row)*QKVD + h*DHEAD;
        att[base + lr]      = (bf16)(acc0[r]*inv);
        att[base + 16 + lr] = (bf16)(acc1[r]*inv);
    }
}

// ---------------------------------------------------------------------------
// Kernel 3: output projection [16384,192] x [192,384] + bias -> f32 d_out.
// Same 16x64 strip blocking as the QKV GEMM: 6 K-steps x 4 WMMAs per wave.
// ---------------------------------------------------------------------------
__global__ void __launch_bounds__(256)
proj_gemm(const bf16* __restrict__ att, const bf16* __restrict__ w2t,
          const float* __restrict__ proj_b, float* __restrict__ out) {
    const int lane = threadIdx.x & 31;
    const int wid  = threadIdx.x >> 5;
    const int tile = blockIdx.x*8 + wid;       // 1024 M-tiles * 6 N-groups
    const int mt = tile / 6, ng = tile % 6;
    const int g = lane >> 4, lr = lane & 15;

    const bf16* arow  = att + (size_t)(mt*16 + lr)*QKVD;
    const bf16* brow0 = w2t + (size_t)(ng*64 + lr)*QKVD;

    v8f acc[4] = {};
    #pragma unroll
    for (int kt = 0; kt < QKVD; kt += 32) {
        v16bf A = cat8(*(const v8bf*)(arow + kt + 16*g),
                       *(const v8bf*)(arow + kt + 16*g + 8));
        #pragma unroll
        for (int j = 0; j < 4; ++j) {
            const bf16* br = brow0 + (size_t)j*16*QKVD;
            v16bf B = cat8(*(const v8bf*)(br + kt + 16*g),
                           *(const v8bf*)(br + kt + 16*g + 8));
            acc[j] = wmma_bf16(A, B, acc[j]);
        }
    }
    #pragma unroll
    for (int j = 0; j < 4; ++j) {
        const int n = ng*64 + j*16 + lr;
        const float bias = proj_b[n];
        #pragma unroll
        for (int r = 0; r < 8; ++r) {
            int m = mt*16 + r + 8*g;
            out[(size_t)m*EMBED + n] = acc[j][r] + bias;
        }
    }
}

// ---------------------------------------------------------------------------
extern "C" void kernel_launch(void* const* d_in, const int* in_sizes, int n_in,
                              void* d_out, int out_size, void* d_ws, size_t ws_size,
                              hipStream_t stream) {
    (void)in_sizes; (void)n_in; (void)out_size; (void)ws_size;
    const float* x      = (const float*)d_in[0];
    const float* qkv_w  = (const float*)d_in[1];
    const float* qkv_b  = (const float*)d_in[2];
    const float* proj_w = (const float*)d_in[3];
    const float* proj_b = (const float*)d_in[4];
    float* out = (float*)d_out;

    // workspace carve-up (~38 MB total)
    char* ws = (char*)d_ws;
    size_t off = 0;
    auto carve = [&](size_t bytes) -> void* {
        void* p = ws + off;
        off = (off + bytes + 255) & ~(size_t)255;
        return p;
    };
    bf16* xb  = (bf16*)carve((size_t)MROWS*EMBED*sizeof(bf16));
    bf16* qb  = (bf16*)carve((size_t)BATCH*NHEAD*SEQ*DHEAD*sizeof(bf16));
    bf16* kb  = (bf16*)carve((size_t)BATCH*NHEAD*SEQ*DHEAD*sizeof(bf16));
    bf16* vtb = (bf16*)carve((size_t)BATCH*NHEAD*DHEAD*SEQ*sizeof(bf16));
    bf16* att = (bf16*)carve((size_t)MROWS*QKVD*sizeof(bf16));
    bf16* w1t = (bf16*)carve((size_t)(3*QKVD)*EMBED*sizeof(bf16));
    bf16* w2t = (bf16*)carve((size_t)EMBED*QKVD*sizeof(bf16));

    conv_x      <<<6144, 256, 0, stream>>>(x, xb);          // 16384*384/4 thr
    prep_weights<<<1152, 256, 0, stream>>>(qkv_w, proj_w, w1t, w2t);
    qkv_gemm    <<<1152, 256, 0, stream>>>(xb, qkv_b, w1t, qb, kb, vtb);
    flash_attn  <<<768,  256, 0, stream>>>(qb, kb, vtb, att);
    proj_gemm   <<<768,  256, 0, stream>>>(att, w2t, proj_b, out);
}
